// Attention_12592844112262
// MI455X (gfx1250) — matrix operand order repaired and checked by reading, not compiled, not run on hardware
//
#include <hip/hip_runtime.h>
#include <hip/hip_bf16.h>
#include <float.h>

// Problem constants (match reference)
static constexpr int B_    = 2;
static constexpr int NQ    = 512;
static constexpr int NC    = 16384;
static constexpr int HEADS = 8;
static constexpr int DHEAD = 64;
static constexpr int INNER = 512;   // HEADS*DHEAD == QDIM == CDIM

typedef __attribute__((ext_vector_type(16))) __bf16 v16bf;
typedef __attribute__((ext_vector_type(8)))  __bf16 v8bf;
typedef __attribute__((ext_vector_type(4)))  __bf16 v4bf;
typedef __attribute__((ext_vector_type(8)))  float  v8f;

#define DEVINL static __device__ __forceinline__

DEVINL v16bf join16(v8bf lo, v8bf hi) {
  v16bf r;
#pragma unroll
  for (int i = 0; i < 8; ++i) { r[i] = lo[i]; r[i + 8] = hi[i]; }
  return r;
}

DEVINL v8f wmma_bf16(v16bf a, v16bf b, v8f c) {
  // D = A(16x32 bf16) * B(32x16 bf16) + C(16x16 f32)
  return __builtin_amdgcn_wmma_f32_16x16x32_bf16(false, a, false, b, (short)0, c,
                                                 false, false);
}

// gfx1250 async global->LDS copy, 16B per lane. lds_off is the byte offset
// within the workgroup LDS allocation (= low 32 bits of the flat shared addr).
DEVINL void async_b128(unsigned lds_off, const __bf16* g) {
  asm volatile("global_load_async_to_lds_b128 %0, %1, off"
               :: "v"(lds_off), "v"(g) : "memory");
}

// ---------------------------------------------------------------------------
// One-time conversion prepasses (each element converted exactly once).
// ---------------------------------------------------------------------------
__global__ __launch_bounds__(256)
void cvt_f32_to_bf16(const float* __restrict__ in, __bf16* __restrict__ out, int n4)
{
  int i = blockIdx.x * 256 + threadIdx.x;
  if (i >= n4) return;
  float4 f = ((const float4*)in)[i];
  v4bf r;
  r[0] = (__bf16)f.x; r[1] = (__bf16)f.y; r[2] = (__bf16)f.z; r[3] = (__bf16)f.w;
  ((v4bf*)out)[i] = r;
}

// W (K x N) f32 -> WT (N x K) bf16.  Tiny matrices; simple is fine.
__global__ __launch_bounds__(256)
void cvt_transpose_bf16(const float* __restrict__ W, __bf16* __restrict__ WT,
                        int Kdim, int N)
{
  int n = blockIdx.x * 32 + (threadIdx.x & 31);
  int k = blockIdx.y * 8  + (threadIdx.x >> 5);
  if (n < N && k < Kdim)
    WT[(size_t)n * Kdim + k] = (__bf16)W[(size_t)k * N + n];
}

// ---------------------------------------------------------------------------
// bf16 GEMM: C = A(bf16, MxK row-major) @ WT(bf16, NxK row-major)^T, fp32 acc.
// K fixed at 512.  Block: 256 threads = 8 waves; block tile 128x64; wave 32x32.
// Inner loop is pure b128 loads + 4 WMMAs.
// mode 0: bf16 row-major out; mode 1: KV epilogue (K row-major + V transposed);
// mode 2: f32 + bias.  All shapes divide the tiles exactly.
// ---------------------------------------------------------------------------
__global__ __launch_bounds__(256)
void gemm_bf16_wmma(const __bf16* __restrict__ A, const __bf16* __restrict__ WT,
                    int ldo, int mode, const float* __restrict__ bias,
                    __bf16* __restrict__ out_bf,
                    __bf16* __restrict__ out_k,
                    __bf16* __restrict__ out_vt,
                    float*  __restrict__ out_f)
{
  constexpr int K = 512;
  const int lane  = threadIdx.x & 31;
  const int wave  = threadIdx.x >> 5;
  const int m0    = blockIdx.x * 128 + (wave & 3) * 32;
  const int n0    = blockIdx.y * 64  + (wave >> 2) * 32;
  const int ln    = lane & 15;
  const int khalf = (lane & 16) >> 1;  // 0 / 8   (unused: A also uses kbase)
  const int kbase = (lane & 16);       // 0 / 16  (B-fragment K offset)

  const __bf16* a0 = A  + (size_t)(m0 + ln) * K;
  const __bf16* a1 = a0 + (size_t)16 * K;
  const __bf16* b0 = WT + (size_t)(n0 + ln) * K + kbase;
  const __bf16* b1 = b0 + (size_t)16 * K;

  v8f acc[2][2] = {};
#pragma unroll 4
  for (int k0 = 0; k0 < K; k0 += 32) {
    v16bf af0 = join16(*(const v8bf*)(a0 + k0 + kbase),
                       *(const v8bf*)(a0 + k0 + kbase + 8));
    v16bf af1 = join16(*(const v8bf*)(a1 + k0 + kbase),
                       *(const v8bf*)(a1 + k0 + kbase + 8));
    v16bf bf0 = *(const v16bf*)(b0 + k0);
    v16bf bf1 = *(const v16bf*)(b1 + k0);
    acc[0][0] = wmma_bf16(af0, bf0, acc[0][0]);
    acc[0][1] = wmma_bf16(af0, bf1, acc[0][1]);
    acc[1][0] = wmma_bf16(af1, bf0, acc[1][0]);
    acc[1][1] = wmma_bf16(af1, bf1, acc[1][1]);
  }

  const int rofs = (lane & 16) >> 1;   // C layout: upper half-wave holds M+8
#pragma unroll
  for (int r = 0; r < 2; ++r) {
#pragma unroll
    for (int s = 0; s < 2; ++s) {
#pragma unroll
      for (int v = 0; v < 8; ++v) {
        int row = m0 + r * 16 + v + rofs;
        int col = n0 + s * 16 + ln;
        float val = acc[r][s][v];
        if (mode == 0) {
          out_bf[(size_t)row * ldo + col] = (__bf16)val;
        } else if (mode == 1) {
          int b = row >> 14;            // / NC (16384)
          int j = row & (NC - 1);
          if (col < INNER)
            out_k[((size_t)b * NC + j) * INNER + col] = (__bf16)val;
          else
            out_vt[((size_t)b * INNER + (col - INNER)) * NC + j] = (__bf16)val;
        } else {
          out_f[(size_t)row * ldo + col] = val + bias[col];
        }
      }
    }
  }
}

// ---------------------------------------------------------------------------
// Flash attention.  Block = 128 threads = 4 waves, all on the same (b,h):
// wave w handles q-tile (qg*4 + w).  K/V 32-token tiles are staged into LDS
// once per block with async global->LDS b128 copies (double-buffered,
// s_wait_asynccnt + two-barrier protocol), cutting L2 traffic 4x vs
// per-wave streaming.  Logits computed transposed (M=j, N=i) so exp'd
// probabilities land directly in the A-fragment layout for the PV WMMA.
// ---------------------------------------------------------------------------
__global__ __launch_bounds__(128)
void flash_attn_wmma(const __bf16* __restrict__ q,
                     const __bf16* __restrict__ k,
                     const __bf16* __restrict__ vt,      // (b, INNER, NC)
                     const unsigned char* __restrict__ mask,  // (b, NC) bool
                     __bf16* __restrict__ out)           // (b, NQ, INNER)
{
  // LDS: K tile [2][32][64] bf16 (8KB) + V^T tile [2][64][32] bf16 (8KB)
  __shared__ __bf16 smem[8192];

  const int tid  = threadIdx.x;           // 0..127
  const int lane = tid & 31;
  const int wave = tid >> 5;              // 0..3
  const int bb   = blockIdx.x;            // 0..127
  const int qt   = (bb & 7) * 4 + wave;   // q-tile 0..31
  const int h    = (bb >> 3) & 7;
  const int b    = bb >> 6;

  const int ln    = lane & 15;
  const int khalf = (lane & 16) >> 1;  // 0/8
  const int kbase = (lane & 16);       // 0/16
  const float scale = 0.125f;          // DHEAD^-0.5

  const __bf16* kb = k  + (size_t)b * NC * INNER + h * DHEAD;
  const __bf16* vb = vt + ((size_t)b * INNER + h * DHEAD) * NC;
  const unsigned char* mb = mask + (size_t)b * NC;

  // ---- per-thread async staging assignment (16 bf16 of K, 16 of V) ----
  const unsigned lds_base = (unsigned)(uintptr_t)(void*)smem;
  const int kj = tid >> 2, kd = (tid & 3) * 16;   // K tile: row kj, d kd..kd+15
  const int vd = tid >> 1, vj = (tid & 1) * 16;   // V tile: row vd, j vj..vj+15
  const unsigned sk_off = lds_base + (unsigned)(kj * 64 + kd) * 2;
  const unsigned sv_off = lds_base + 8192u + (unsigned)(vd * 32 + vj) * 2;
  const __bf16* kg = kb + (size_t)kj * INNER + kd;
  const __bf16* vg = vb + (size_t)vd * NC + vj;

  // Q as B-fragments (d-chunks 0..31, 32..63), resident for whole pass.
  const __bf16* qrow = q + ((size_t)(b * NQ + qt * 16 + ln) * INNER + h * DHEAD);
  v16bf bq[2];
#pragma unroll
  for (int c = 0; c < 2; ++c)
    bq[c] = join16(*(const v8bf*)(qrow + c * 32 + khalf),
                   *(const v8bf*)(qrow + c * 32 + 16 + khalf));

  v8f o[4] = {};                       // O: 16 queries x 64 d (4 chunks)
  float mrow = -FLT_MAX, lrow = 0.f;

  // prologue: stage tile 0 into buffer 0
  async_b128(sk_off,      kg);
  async_b128(sk_off + 16, kg + 8);
  async_b128(sv_off,      vg);
  async_b128(sv_off + 16, vg + 8);

  constexpr int NSTEPS = NC / 32;
  for (int it = 0; it < NSTEPS; ++it) {
    const int buf = it & 1;
    const int j0  = it * 32;
    // prefetch next tile into the other buffer (its readers passed the
    // trailing barrier of iteration it-1, so overwrite is safe)
    if (it + 1 < NSTEPS) {
      const unsigned bo = (buf ^ 1) * 4096u;
      const __bf16* kgp = kg + (size_t)(j0 + 32) * INNER;
      const __bf16* vgp = vg + (j0 + 32);
      async_b128(sk_off + bo,      kgp);
      async_b128(sk_off + bo + 16, kgp + 8);
      async_b128(sv_off + bo,      vgp);
      async_b128(sv_off + bo + 16, vgp + 8);
      asm volatile("s_wait_asynccnt 0x4" ::: "memory");  // this buf's 4 done
    } else {
      asm volatile("s_wait_asynccnt 0x0" ::: "memory");
    }
    __syncthreads();   // all waves' writes to `buf` visible

    const __bf16* skb = smem + buf * 2048;          // [32][64]
    const __bf16* svb = smem + 4096 + buf * 2048;   // [64][32]

    // ---- logits: sim^T[j][i], two 16x16 tiles over j, K-dim = d (64) ----
    v8f c0 = {}, c1 = {};
#pragma unroll
    for (int ch = 0; ch < 2; ++ch) {
      const __bf16* kr0 = skb + (size_t)ln * 64 + ch * 32;
      const __bf16* kr1 = skb + (size_t)(16 + ln) * 64 + ch * 32;
      v16bf ak0 = join16(*(const v8bf*)(kr0 + khalf), *(const v8bf*)(kr0 + 16 + khalf));
      v16bf ak1 = join16(*(const v8bf*)(kr1 + khalf), *(const v8bf*)(kr1 + 16 + khalf));
      c0 = wmma_bf16(ak0, bq[ch], c0);
      c1 = wmma_bf16(ak1, bq[ch], c1);
    }

    // ---- scale + mask (per lane: query i=ln; j = v + khalf (+16)) ----
    unsigned long long mm0 = *(const unsigned long long*)(mb + j0 + khalf);
    unsigned long long mm1 = *(const unsigned long long*)(mb + j0 + 16 + khalf);
    float p0[8], p1[8];
    float mloc = -FLT_MAX;
#pragma unroll
    for (int v = 0; v < 8; ++v) {
      float s0 = c0[v] * scale;
      float s1 = c1[v] * scale;
      s0 = ((mm0 >> (8 * v)) & 0xffull) ? s0 : -FLT_MAX;
      s1 = ((mm1 >> (8 * v)) & 0xffull) ? s1 : -FLT_MAX;
      p0[v] = s0; p1[v] = s1;
      mloc = fmaxf(mloc, fmaxf(s0, s1));
    }
    // combine the two j-halves of each query (lane i <-> lane i+16)
    mloc = fmaxf(mloc, __shfl_xor(mloc, 16, 32));
    float mnew  = fmaxf(mrow, mloc);
    float alpha = __expf(mrow - mnew);
    float psum = 0.f;
#pragma unroll
    for (int v = 0; v < 8; ++v) {
      p0[v] = __expf(p0[v] - mnew);
      p1[v] = __expf(p1[v] - mnew);
      psum += p0[v] + p1[v];
    }
    psum += __shfl_xor(psum, 16, 32);
    lrow = lrow * alpha + psum;
    mrow = mnew;

    // ---- rescale O: row of acc VGPR v is query (v + khalf) ----
#pragma unroll
    for (int v = 0; v < 8; ++v) {
      float av = __shfl(alpha, v + khalf, 32);
      o[0][v] *= av; o[1][v] *= av; o[2][v] *= av; o[3][v] *= av;
    }

    // ---- P already matches the A-fragment layout exactly ----
    v16bf ap;
#pragma unroll
    for (int v = 0; v < 8; ++v) { ap[v] = (__bf16)p0[v]; ap[v + 8] = (__bf16)p1[v]; }

    // ---- O += P @ V  (V^T in LDS: contiguous over j per d-row) ----
#pragma unroll
    for (int ch = 0; ch < 4; ++ch) {
      const __bf16* vr = svb + (size_t)(ch * 16 + ln) * 32;
      v16bf bv = join16(*(const v8bf*)(vr + khalf), *(const v8bf*)(vr + 16 + khalf));
      o[ch] = wmma_bf16(ap, bv, o[ch]);
    }

    __syncthreads();   // everyone done reading `buf` before it is re-staged
  }

  // ---- finalize: divide by row sums, store bf16 ----
#pragma unroll
  for (int v = 0; v < 8; ++v) {
    float inv = 1.0f / __shfl(lrow, v + khalf, 32);
    int i = qt * 16 + v + khalf;
    __bf16* orow = out + ((size_t)(b * NQ + i) * INNER + h * DHEAD);
#pragma unroll
    for (int ch = 0; ch < 4; ++ch)
      orow[ch * 16 + ln] = (__bf16)(o[ch][v] * inv);
  }
}

// ---------------------------------------------------------------------------
extern "C" void kernel_launch(void* const* d_in, const int* in_sizes, int n_in,
                              void* d_out, int out_size, void* d_ws, size_t ws_size,
                              hipStream_t stream) {
  const float*         x    = (const float*)d_in[0];          // (2,512,512)
  const float*         ctx  = (const float*)d_in[1];          // (2,16384,512)
  const unsigned char* mask = (const unsigned char*)d_in[2];  // (2,16384) bool
  const float*         Wq   = (const float*)d_in[3];          // (512,512)
  const float*         Wkv  = (const float*)d_in[4];          // (512,1024)
  const float*         Wout = (const float*)d_in[5];          // (512,512)
  const float*         bout = (const float*)d_in[6];          // (512,)
  float* out = (float*)d_out;                                 // (2,512,512)

  // Workspace layout (bf16 elements; ~107 MB total)
  __bf16* xb   = (__bf16*)d_ws;                               // (2,512,512)
  __bf16* cb   = xb   + (size_t)B_ * NQ * INNER;              // (2,16384,512)
  __bf16* wqt  = cb   + (size_t)B_ * NC * INNER;              // (512,512) N x K
  __bf16* wkvt = wqt  + (size_t)INNER * INNER;                // (1024,512) N x K
  __bf16* wot  = wkvt + (size_t)2 * INNER * INNER;            // (512,512) N x K
  __bf16* qws  = wot  + (size_t)INNER * INNER;                // (2,512,512)
  __bf16* kws  = qws  + (size_t)B_ * NQ * INNER;              // (2,16384,512)
  __bf16* vtws = kws  + (size_t)B_ * NC * INNER;              // (2,512,16384)
  __bf16* awsb = vtws + (size_t)B_ * NC * INNER;              // (2,512,512)

  dim3 blk(256);

  // 0) one-time f32 -> bf16 conversions / weight transposes
  cvt_f32_to_bf16<<<dim3((B_ * NQ * INNER / 4) / 256), blk, 0, stream>>>(
      x, xb, B_ * NQ * INNER / 4);
  cvt_f32_to_bf16<<<dim3((B_ * NC * INNER / 4) / 256), blk, 0, stream>>>(
      ctx, cb, B_ * NC * INNER / 4);
  cvt_transpose_bf16<<<dim3(INNER / 32, INNER / 8), blk, 0, stream>>>(Wq, wqt, INNER, INNER);
  cvt_transpose_bf16<<<dim3(2 * INNER / 32, INNER / 8), blk, 0, stream>>>(Wkv, wkvt, INNER, 2 * INNER);
  cvt_transpose_bf16<<<dim3(INNER / 32, INNER / 8), blk, 0, stream>>>(Wout, wot, INNER, INNER);

  // 1) Q projection: (1024x512) @ (512x512) -> bf16
  gemm_bf16_wmma<<<dim3(1024 / 128, 512 / 64), blk, 0, stream>>>(
      xb, wqt, 512, 0, nullptr, qws, nullptr, nullptr, nullptr);

  // 2) KV projection: (32768x512) @ (512x1024) -> K row-major, V transposed
  gemm_bf16_wmma<<<dim3(32768 / 128, 1024 / 64), blk, 0, stream>>>(
      cb, wkvt, 0, 1, nullptr, nullptr, kws, vtws, nullptr);

  // 3) flash attention: 128 blocks x 4 waves, K/V shared via async LDS tiles
  flash_attn_wmma<<<dim3(128), dim3(128), 0, stream>>>(qws, kws, vtws, mask, awsb);

  // 4) output projection + bias: (1024x512) @ (512x512) -> f32
  gemm_bf16_wmma<<<dim3(1024 / 128, 512 / 64), blk, 0, stream>>>(
      awsb, wot, 512, 2, bout, nullptr, nullptr, nullptr, out);
}
